// Ernie4_5_Attention_6708738916444
// MI455X (gfx1250) — matrix operand order repaired and checked by reading, not compiled, not run on hardware
//
#include <hip/hip_runtime.h>
#include <hip/hip_bf16.h>
#include <math.h>

// ---------------- problem constants ----------------
constexpr int HIDDEN   = 2048;
constexpr int N_HEADS  = 16;
constexpr int N_KV     = 4;
constexpr int HEAD_DIM = 128;
constexpr int Bq       = 2;
constexpr int Sq       = 2048;
constexpr int BS       = Bq * Sq;              // 4096 rows
constexpr int QDIM     = N_HEADS * HEAD_DIM;   // 2048
constexpr int KVDIM    = N_KV * HEAD_DIM;      // 512
constexpr float LOG_ROPE_BASE = 9.210340371976184f; // ln(10000)
constexpr float SCALE  = 0.08838834764831845f;      // 1/sqrt(128)

// ---------------- vector types ----------------
typedef __attribute__((ext_vector_type(16))) __bf16 v16bf;
typedef __attribute__((ext_vector_type(8)))  __bf16 bf16x8;
typedef __attribute__((ext_vector_type(4)))  __bf16 bf16x4;
typedef __attribute__((ext_vector_type(8)))  float  v8f;

union V16 { v16bf v; bf16x8 h[2]; };

__device__ __forceinline__ v8f wmma_bf16(v16bf a, v16bf b, v8f c) {
  // D(16x16,f32) = A(16x32,bf16) x B(32x16,bf16) + C
  return __builtin_amdgcn_wmma_f32_16x16x32_bf16(false, a, false, b,
                                                 (short)0, c, false, false);
}

// ---- CDNA5 async memory->LDS copy (16B per lane, ASYNCcnt-tracked) ----
__device__ __forceinline__ void async_copy_b128_to_lds(__bf16* lptr,
                                                       const __bf16* gptr) {
  // flat LDS aperture keeps the LDS offset in addr[31:0] (ISA 10.2)
  unsigned lds = (unsigned)(size_t)lptr;
  unsigned long long ga = (unsigned long long)(size_t)gptr;
  asm volatile("global_load_async_to_lds_b128 %0, %1, off"
               :: "v"(lds), "v"(ga) : "memory");
}
__device__ __forceinline__ void wait_async0() {
  asm volatile("s_wait_asynccnt 0x0" ::: "memory");
}

// ---------------- fp32 -> bf16 convert ----------------
__global__ void cvt_bf16_x4(const float4* __restrict__ src,
                            bf16x4* __restrict__ dst, int n4) {
  int i = blockIdx.x * blockDim.x + threadIdx.x;
  if (i < n4) {
    float4 f = src[i];
    bf16x4 o;
    o[0] = (__bf16)f.x; o[1] = (__bf16)f.y;
    o[2] = (__bf16)f.z; o[3] = (__bf16)f.w;
    dst[i] = o;
  }
}

// ---------------- WMMA GEMM: Y[row,col] = sum_k X[row,k] * W[col,k] ----------
// X: (BS x K) bf16 row-major.  W: (N x K) bf16 row-major (i.e. Y = X @ W^T).
// Optional fused interleaved RoPE on the output (d = col & 127, pos = row % S).
template<bool ROPE, bool OUT_F32>
__global__ void __launch_bounds__(256, 1)
gemm_wmma_bf16(const __bf16* __restrict__ X,
               const __bf16* __restrict__ W,
               __bf16* __restrict__ Yb,
               float*  __restrict__ Yf,
               int N, int K) {
  const int lane = threadIdx.x & 31;
  const int wave = threadIdx.x >> 5;
  const int half = lane >> 4;
  const int l15  = lane & 15;
  const int wm   = wave >> 1;          // 0..3
  const int wn   = wave & 1;           // 0..1
  const int rowBase = blockIdx.y * 128 + wm * 32;
  const int colBase = blockIdx.x * 128 + wn * 64;

  v8f acc[2][4];
#pragma unroll
  for (int mf = 0; mf < 2; ++mf)
#pragma unroll
    for (int nf = 0; nf < 4; ++nf) acc[mf][nf] = (v8f){};

  for (int k0 = 0; k0 < K; k0 += 32) {
    V16 Af[2];
#pragma unroll
    for (int mf = 0; mf < 2; ++mf) {
      const __bf16* p = X + (size_t)(rowBase + mf * 16 + l15) * K + k0 + half * 16;
      Af[mf].h[0] = *(const bf16x8*)p;
      Af[mf].h[1] = *(const bf16x8*)(p + 8);
      __builtin_prefetch(p + 32, 0, 3);          // next k-step A stream (WGP)
    }
    V16 Bf[4];
#pragma unroll
    for (int nf = 0; nf < 4; ++nf) {
      const __bf16* p = W + (size_t)(colBase + nf * 16 + l15) * K + k0 + half * 16;
      Bf[nf].h[0] = *(const bf16x8*)p;
      Bf[nf].h[1] = *(const bf16x8*)(p + 8);
      __builtin_prefetch(p + 32, 0, 3);          // next k-step B stream (WGP)
    }
#pragma unroll
    for (int mf = 0; mf < 2; ++mf)
#pragma unroll
      for (int nf = 0; nf < 4; ++nf)
        acc[mf][nf] = wmma_bf16(Af[mf].v, Bf[nf].v, acc[mf][nf]);
  }

  // epilogue: D-fragment lane holds col n = l15, rows m = half*8 + r
#pragma unroll
  for (int mf = 0; mf < 2; ++mf) {
#pragma unroll
    for (int nf = 0; nf < 4; ++nf) {
      const int col = colBase + nf * 16 + l15;
      float invf = 0.f;
      if (ROPE) {
        const int d = col & (HEAD_DIM - 1);
        invf = __expf(-(float)(2 * (d >> 1)) * (LOG_ROPE_BASE / (float)HEAD_DIM));
      }
#pragma unroll
      for (int r = 0; r < 8; ++r) {
        const int row = rowBase + mf * 16 + half * 8 + r;
        float v = acc[mf][nf][r];
        if (ROPE) {
          const int d = col & (HEAD_DIM - 1);
          const float partner = __shfl_xor(v, 1, 32);  // (even,odd) pair lane
          float s, c;
          __sincosf((float)(row & (Sq - 1)) * invf, &s, &c);
          v = v * c + ((d & 1) ? partner : -partner) * s;
        }
        if (OUT_F32) Yf[(size_t)row * N + col] = v;
        else         Yb[(size_t)row * N + col] = (__bf16)v;
      }
    }
  }
}

// ---------------- flash attention (causal, GQA 16q/4kv heads) ---------------
// grid = (S/128, N_HEADS, B); 256 threads = 8 waves, wave w owns q rows
// [q0+16w, q0+16w+16). Key blocks of 32, online softmax, WMMA for QK^T and PV.
__global__ void __launch_bounds__(256, 1)
flash_attn_wmma(const __bf16* __restrict__ Qb,
                const __bf16* __restrict__ Kb,
                const __bf16* __restrict__ Vb,
                __bf16* __restrict__ Ob) {
  const int q0   = blockIdx.x * 128;
  const int hq   = blockIdx.y;
  const int b    = blockIdx.z;
  const int hkv  = hq >> 2;            // 16 -> 4 GQA
  const int lane = threadIdx.x & 31;
  const int wave = threadIdx.x >> 5;
  const int half = lane >> 4;
  const int l15  = lane & 15;
  const int qw   = q0 + wave * 16;     // sequence-local first q row of wave

  __shared__ __bf16 Kt[32 * 128];      // [key][dim]     row-major
  __shared__ __bf16 Vt[128 * 40];      // [dim][key]     transposed, padded
  __shared__ __bf16 Pt[8 * 16 * 40];   // per-wave P transpose scratch

  // Q fragments: 16 rows x 128 dims = 4 A-fragments, kept in registers
  V16 Qf[4];
  {
    const __bf16* qp = Qb + (size_t)(b * Sq + qw + l15) * QDIM + hq * HEAD_DIM;
#pragma unroll
    for (int f = 0; f < 4; ++f) {
      Qf[f].h[0] = *(const bf16x8*)(qp + f * 32 + half * 16);
      Qf[f].h[1] = *(const bf16x8*)(qp + f * 32 + half * 16 + 8);
    }
  }

  v8f O[8];
#pragma unroll
  for (int nb = 0; nb < 8; ++nb) O[nb] = (v8f){};
  float mrow[8], lrow[8];
#pragma unroll
  for (int r = 0; r < 8; ++r) { mrow[r] = -1e30f; lrow[r] = 0.f; }

  const int nkb = (q0 + 128) / 32;     // uniform per workgroup (causal bound)
  for (int kb = 0; kb < nkb; ++kb) {
    const int k0 = kb * 32;
    __syncthreads();
    // cooperative K/V tile load: thread t -> key row t/8, 16-elem segment t%8
    {
      const int krow = threadIdx.x >> 3;
      const int seg  = threadIdx.x & 7;
      // K tile: CDNA5 async DMA straight into LDS (no VGPR round-trip)
      const __bf16* kp = Kb + (size_t)(b * Sq + k0 + krow) * KVDIM +
                         hkv * HEAD_DIM + seg * 16;
      async_copy_b128_to_lds(&Kt[krow * 128 + seg * 16],     kp);
      async_copy_b128_to_lds(&Kt[krow * 128 + seg * 16 + 8], kp + 8);
      // V tile: needs transpose -> manual scatter into LDS
      const __bf16* vp = Vb + (size_t)(b * Sq + k0 + krow) * KVDIM +
                         hkv * HEAD_DIM + seg * 16;
      bf16x8 v0 = *(const bf16x8*)vp;
      bf16x8 v1 = *(const bf16x8*)(vp + 8);
#pragma unroll
      for (int j = 0; j < 8; ++j) {
        Vt[(seg * 16 + j) * 40 + krow]     = v0[j];
        Vt[(seg * 16 + 8 + j) * 40 + krow] = v1[j];
      }
      // prefetch next key-block K/V streams into near cache
      __builtin_prefetch(kp + 32 * KVDIM, 0, 3);
      __builtin_prefetch(vp + 32 * KVDIM, 0, 3);
    }
    wait_async0();                      // K-tile async DMA complete
    __syncthreads();

    // scores S(16x32) = Q(16x128) @ K^T(128x32): two 16-key fragments
    v8f Sf[2] = {(v8f){}, (v8f){}};
#pragma unroll
    for (int c = 0; c < 2; ++c) {
#pragma unroll
      for (int f = 0; f < 4; ++f) {
        V16 Bf;
        const __bf16* kp = &Kt[(c * 16 + l15) * 128 + f * 32 + half * 16];
        Bf.h[0] = *(const bf16x8*)kp;
        Bf.h[1] = *(const bf16x8*)(kp + 8);
        Sf[c] = wmma_bf16(Qf[f].v, Bf.v, Sf[c]);
      }
    }

    // scale + causal mask + online softmax
#pragma unroll
    for (int r = 0; r < 8; ++r) {
      const int qrow = qw + half * 8 + r;     // sequence-local query index
      float a0 = Sf[0][r] * SCALE;
      float a1 = Sf[1][r] * SCALE;
      if (k0 + l15 > qrow)      a0 = -1e30f;
      if (k0 + 16 + l15 > qrow) a1 = -1e30f;
      float mx = fmaxf(a0, a1);
#pragma unroll
      for (int off = 1; off < 16; off <<= 1)
        mx = fmaxf(mx, __shfl_xor(mx, off, 32));
      const float mnew  = fmaxf(mrow[r], mx);
      const float alpha = __expf(mrow[r] - mnew);
      const float p0 = __expf(a0 - mnew);
      const float p1 = __expf(a1 - mnew);
      float ps = p0 + p1;
#pragma unroll
      for (int off = 1; off < 16; off <<= 1)
        ps += __shfl_xor(ps, off, 32);
      lrow[r] = lrow[r] * alpha + ps;
      mrow[r] = mnew;
#pragma unroll
      for (int nb = 0; nb < 8; ++nb) O[nb][r] *= alpha;
      // stash P (D-layout) to LDS for transpose into A-layout
      const int m = half * 8 + r;
      Pt[wave * 640 + m * 40 + l15]      = (__bf16)p0;
      Pt[wave * 640 + m * 40 + 16 + l15] = (__bf16)p1;
    }

    // P as A-fragment (16x32 bf16)
    V16 Pf;
    {
      const __bf16* pp = &Pt[wave * 640 + l15 * 40];
      Pf.h[0] = *(const bf16x8*)(pp + half * 16);
      Pf.h[1] = *(const bf16x8*)(pp + half * 16 + 8);
    }
    // O(16x128) += P(16x32) @ V(32x128): 8 dim-blocks of 16
#pragma unroll
    for (int nb = 0; nb < 8; ++nb) {
      V16 Vf;
      const __bf16* vv = &Vt[(nb * 16 + l15) * 40 + half * 16];
      Vf.h[0] = *(const bf16x8*)vv;
      Vf.h[1] = *(const bf16x8*)(vv + 8);
      O[nb] = wmma_bf16(Pf.v, Vf.v, O[nb]);
    }
  }

  // normalize and store attn (bf16, layout [b*S+s][h*128+d])
#pragma unroll
  for (int r = 0; r < 8; ++r) {
    const float inv = lrow[r] > 0.f ? 1.f / lrow[r] : 0.f;
    const size_t row = (size_t)(b * Sq + qw + half * 8 + r);
#pragma unroll
    for (int nb = 0; nb < 8; ++nb)
      Ob[row * QDIM + hq * HEAD_DIM + nb * 16 + l15] = (__bf16)(O[nb][r] * inv);
  }
}

// ---------------- host launcher ----------------
extern "C" void kernel_launch(void* const* d_in, const int* in_sizes, int n_in,
                              void* d_out, int out_size, void* d_ws, size_t ws_size,
                              hipStream_t stream) {
  (void)in_sizes; (void)n_in; (void)out_size; (void)ws_size;
  const float* x  = (const float*)d_in[0];
  const float* Wq = (const float*)d_in[1];
  const float* Wk = (const float*)d_in[2];
  const float* Wv = (const float*)d_in[3];
  const float* Wo = (const float*)d_in[4];
  float* out = (float*)d_out;

  // workspace carve-up (bf16 buffers), 256B aligned
  char* ws = (char*)d_ws;
  size_t off = 0;
  auto carve = [&](size_t bytes) {
    void* p = ws + off;
    off += (bytes + 255) & ~(size_t)255;
    return p;
  };
  __bf16* xb   = (__bf16*)carve((size_t)BS * HIDDEN * 2);
  __bf16* Wqb  = (__bf16*)carve((size_t)QDIM * HIDDEN * 2);
  __bf16* Wkb  = (__bf16*)carve((size_t)KVDIM * HIDDEN * 2);
  __bf16* Wvb  = (__bf16*)carve((size_t)KVDIM * HIDDEN * 2);
  __bf16* Wob  = (__bf16*)carve((size_t)HIDDEN * QDIM * 2);
  __bf16* Qbuf = (__bf16*)carve((size_t)BS * QDIM * 2);
  __bf16* Kbuf = (__bf16*)carve((size_t)BS * KVDIM * 2);
  __bf16* Vbuf = (__bf16*)carve((size_t)BS * KVDIM * 2);
  __bf16* Abuf = (__bf16*)carve((size_t)BS * QDIM * 2);

  // 1) convert inputs to bf16
  auto cvt = [&](const float* s, __bf16* d, size_t n) {
    int n4 = (int)(n / 4);
    cvt_bf16_x4<<<(n4 + 255) / 256, 256, 0, stream>>>(
        (const float4*)s, (bf16x4*)d, n4);
  };
  cvt(x,  xb,  (size_t)BS * HIDDEN);
  cvt(Wq, Wqb, (size_t)QDIM * HIDDEN);
  cvt(Wk, Wkb, (size_t)KVDIM * HIDDEN);
  cvt(Wv, Wvb, (size_t)KVDIM * HIDDEN);
  cvt(Wo, Wob, (size_t)HIDDEN * QDIM);

  // 2) QKV projections (RoPE fused for Q,K)
  gemm_wmma_bf16<true, false><<<dim3(QDIM / 128, BS / 128), 256, 0, stream>>>(
      xb, Wqb, Qbuf, nullptr, QDIM, HIDDEN);
  gemm_wmma_bf16<true, false><<<dim3(KVDIM / 128, BS / 128), 256, 0, stream>>>(
      xb, Wkb, Kbuf, nullptr, KVDIM, HIDDEN);
  gemm_wmma_bf16<false, false><<<dim3(KVDIM / 128, BS / 128), 256, 0, stream>>>(
      xb, Wvb, Vbuf, nullptr, KVDIM, HIDDEN);

  // 3) causal flash attention
  flash_attn_wmma<<<dim3(Sq / 128, N_HEADS, Bq), 256, 0, stream>>>(
      Qbuf, Kbuf, Vbuf, Abuf);

  // 4) output projection -> fp32
  gemm_wmma_bf16<false, true><<<dim3(HIDDEN / 128, BS / 128), 256, 0, stream>>>(
      Abuf, Wob, nullptr, out, HIDDEN, HIDDEN);
}
